// RNNModel_13657996001526
// MI455X (gfx1250) — matrix operand order repaired and checked by reading, not compiled, not run on hardware
//
#include <hip/hip_runtime.h>

typedef __attribute__((ext_vector_type(16))) _Float16 v16h;
typedef __attribute__((ext_vector_type(8)))  _Float16 v8h;
typedef __attribute__((ext_vector_type(8)))  float    v8f;

#define HID    100   // hidden size
#define G4     400   // 4*HID (gate vector length)
#define TT     512   // timesteps
#define KPAD   128   // hidden dim padded for WMMA K
#define KST    136   // LDS row stride (halves) for f16 tiles (bank-conflict pad)
#define GST    404   // LDS row stride (floats) for gate buffer (bank-conflict pad)
#define MROWS  16    // batch rows per workgroup (one WMMA M tile)
#define NTILES 25    // 400 / 16
#define NWAVES 8
#define TRES   3     // register-resident n-tiles per wave (tiles 0..23)
#define BLOCK  256

// A-matrix fragment (16x32 f16, M=row per lane 0..15):
// lanes 0-15 hold K = kc*32 + {0..7, 16..23}; lanes 16-31 hold K = kc*32 + {8..15, 24..31}
__device__ __forceinline__ v16h ld_a_frag(const _Float16* rowp, int kc, int lane) {
  const int base = kc * 32 + ((lane & 16) ? 8 : 0);
  union { v16h v; v8h h[2]; } u;
  u.h[0] = *(const v8h*)(rowp + base);
  u.h[1] = *(const v8h*)(rowp + base + 16);
  return u.v;
}

// B-matrix fragment (32x16 f16, N=col per lane 0..15):
// lanes 0-15 hold K = kc*32 + {0..7, 16..23}; lanes 16-31 hold K = kc*32 + {8..15, 24..31}
__device__ __forceinline__ v16h ld_b_frag(const _Float16* colp, int kc, int lane) {
  const int base = kc * 32 + ((lane & 16) ? 8 : 0);
  union { v16h v; v8h h[2]; } u;
  u.h[0] = *(const v8h*)(colp + base);
  u.h[1] = *(const v8h*)(colp + base + 16);
  return u.v;
}

__device__ __forceinline__ float sigm(float x)     { return 1.f / (1.f + __expf(-x)); }
__device__ __forceinline__ float tanhfast(float x) { return 1.f - 2.f / (__expf(2.f * x) + 1.f); }

__global__ __launch_bounds__(BLOCK, 1) void lstm_wmma_kernel(
    const float* __restrict__ x,     // [B, TT] (I==1)
    const float* __restrict__ W_ih,  // [400]   (I==1)
    const float* __restrict__ W_hh,  // [400, 100]
    const float* __restrict__ b_ih,  // [400]
    const float* __restrict__ b_hh,  // [400]
    const float* __restrict__ fc_W,  // [100]
    const float* __restrict__ fc_b,  // [1]
    float* __restrict__ out)         // [B]
{
  extern __shared__ char smem[];
  _Float16* Wsh = (_Float16*)smem;                           // [400][KST] f16 W_hh, K padded
  float*    Gsh = (float*)(smem + G4 * KST * 2);             // [16][GST]  gate pre-activations
  _Float16* Hh  = (_Float16*)((char*)Gsh + MROWS * GST * 4); // [16][KST]  h in f16 (A operand)
  float*    Hf  = (float*)((char*)Hh + MROWS * KST * 2);     // [16][100]  h_last f32 (final FC)
  float*    Xsh = Hf + MROWS * HID;                          // [16]       x values for current t

  const int tid  = threadIdx.x;
  const int lane = tid & 31;
  const int wave = tid >> 5;
  const int b0   = blockIdx.x * MROWS;

  // ---------------- setup: stage weights + zero state ----------------
  for (int i = tid; i < G4 * KPAD; i += BLOCK) {
    const int g = i >> 7, k = i & (KPAD - 1);
    Wsh[g * KST + k] = (k < HID) ? (_Float16)W_hh[g * HID + k] : (_Float16)0.f;
  }
  for (int i = tid; i < MROWS * KPAD; i += BLOCK)
    Hh[(i >> 7) * KST + (i & (KPAD - 1))] = (_Float16)0.f;
  const float* xq = x + (size_t)(b0 + (tid & 15)) * TT;  // per-row x pointer (threads 0-15)
  if (tid < MROWS) Xsh[tid] = xq[0];
  __syncthreads();

  const int frow = lane & 15;             // A row / B col / C col within tile
  const int mb   = (lane & 16) ? 8 : 0;   // C/D row base for this lane half

  // ---- resident W_hh B-fragments (tiles 0..23) + per-column input/bias coefs ----
  v16h  breg[TRES][4];
  float cWr[TRES], cBr[TRES];
  #pragma unroll
  for (int i = 0; i < TRES; ++i) {
    const int nt  = wave + i * NWAVES;          // always < 24
    const int col = nt * 16 + frow;
    const _Float16* colp = Wsh + col * KST;
    #pragma unroll
    for (int kc = 0; kc < 4; ++kc) breg[i][kc] = ld_b_frag(colp, kc, lane);
    cWr[i] = W_ih[col];
    cBr[i] = b_ih[col] + b_hh[col];
  }
  // tile 24 (cols 384..399): streamed from LDS by wave 0 each step
  const int  colx = 24 * 16 + frow;
  const _Float16* colpx = Wsh + colx * KST;
  const float cWx = W_ih[colx];
  const float cBx = b_ih[colx] + b_hh[colx];

  // ---- per-thread elementwise mapping: 16 threads per batch row, no division ----
  const int er = tid >> 4;                 // batch row 0..15
  const int ej = tid & 15;                 // hidden base 0..15, step 16
  const _Float16* arow = Hh + frow * KST;
  float creg[7], hreg[7];
  #pragma unroll
  for (int k = 0; k < 7; ++k) { creg[k] = 0.f; hreg[k] = 0.f; }

  #pragma unroll 1
  for (int t = 0; t < TT; ++t) {
    // ---------- phase A: G = (x*W_ih + b) + h @ W_hh^T via WMMA ----------
    const v16h a0 = ld_a_frag(arow, 0, lane);
    const v16h a1 = ld_a_frag(arow, 1, lane);
    const v16h a2 = ld_a_frag(arow, 2, lane);
    const v16h a3 = ld_a_frag(arow, 3, lane);
    float xv[8];
    #pragma unroll
    for (int j = 0; j < 8; ++j) xv[j] = Xsh[mb + j];   // broadcast reads

    #pragma unroll
    for (int i = 0; i < TRES; ++i) {
      const int nt = wave + i * NWAVES;
      v8f acc;
      #pragma unroll
      for (int j = 0; j < 8; ++j) acc[j] = fmaf(xv[j], cWr[i], cBr[i]);
      acc = __builtin_amdgcn_wmma_f32_16x16x32_f16(false, a0, false, breg[i][0],
                                                   (short)0, acc, false, false);
      acc = __builtin_amdgcn_wmma_f32_16x16x32_f16(false, a1, false, breg[i][1],
                                                   (short)0, acc, false, false);
      acc = __builtin_amdgcn_wmma_f32_16x16x32_f16(false, a2, false, breg[i][2],
                                                   (short)0, acc, false, false);
      acc = __builtin_amdgcn_wmma_f32_16x16x32_f16(false, a3, false, breg[i][3],
                                                   (short)0, acc, false, false);
      // C/D layout: VGPR j -> row mb+j, col = nt*16 + frow
      float* gcol = Gsh + nt * 16 + frow;
      #pragma unroll
      for (int j = 0; j < 8; ++j)
        gcol[(mb + j) * GST] = acc[j];
    }

    if (wave == 0) {   // 25th tile: B streamed from LDS (overlaps other waves' WMMAs)
      v8f acc;
      #pragma unroll
      for (int j = 0; j < 8; ++j) acc[j] = fmaf(xv[j], cWx, cBx);
      acc = __builtin_amdgcn_wmma_f32_16x16x32_f16(false, a0, false, ld_b_frag(colpx, 0, lane),
                                                   (short)0, acc, false, false);
      acc = __builtin_amdgcn_wmma_f32_16x16x32_f16(false, a1, false, ld_b_frag(colpx, 1, lane),
                                                   (short)0, acc, false, false);
      acc = __builtin_amdgcn_wmma_f32_16x16x32_f16(false, a2, false, ld_b_frag(colpx, 2, lane),
                                                   (short)0, acc, false, false);
      acc = __builtin_amdgcn_wmma_f32_16x16x32_f16(false, a3, false, ld_b_frag(colpx, 3, lane),
                                                   (short)0, acc, false, false);
      float* gcol = Gsh + 24 * 16 + frow;
      #pragma unroll
      for (int j = 0; j < 8; ++j)
        gcol[(mb + j) * GST] = acc[j];
    }
    __syncthreads();

    // ---------- phase B: elementwise LSTM cell update (state in registers) ----------
    {
      const float* gr = Gsh + er * GST;
      #pragma unroll
      for (int k = 0; k < 7; ++k) {
        const int j = ej + 16 * k;
        if (j < HID) {
          const float gi = gr[j];
          const float gf = gr[HID + j];
          const float gg = gr[2 * HID + j];
          const float go = gr[3 * HID + j];
          const float ig = sigm(gi);
          const float fg = sigm(gf);
          const float og = sigm(go);
          const float gt = tanhfast(gg);
          const float cn = fg * creg[k] + ig * gt;
          const float hn = og * tanhfast(cn);
          creg[k] = cn;
          hreg[k] = hn;
          Hh[er * KST + j] = (_Float16)hn;   // K-pad region [100,128) stays zero
        }
      }
      if (tid < MROWS && t + 1 < TT) Xsh[tid] = xq[t + 1];  // prefetch next step's x
    }
    __syncthreads();
  }

  // ---------------- final FC: out[b] = h_last . fc_W + fc_b ----------------
  #pragma unroll
  for (int k = 0; k < 7; ++k) {
    const int j = ej + 16 * k;
    if (j < HID) Hf[er * HID + j] = hreg[k];
  }
  __syncthreads();
  if (tid < MROWS) {
    float s = fc_b[0];
    const float* hr = Hf + tid * HID;
    #pragma unroll 4
    for (int j = 0; j < HID; ++j) s += hr[j] * fc_W[j];
    out[b0 + tid] = s;
  }
}

extern "C" void kernel_launch(void* const* d_in, const int* in_sizes, int n_in,
                              void* d_out, int out_size, void* d_ws, size_t ws_size,
                              hipStream_t stream) {
  (void)n_in; (void)d_ws; (void)ws_size; (void)out_size;
  const float* x    = (const float*)d_in[0];
  const float* W_ih = (const float*)d_in[1];
  const float* W_hh = (const float*)d_in[2];
  const float* b_ih = (const float*)d_in[3];
  const float* b_hh = (const float*)d_in[4];
  const float* fc_W = (const float*)d_in[5];
  const float* fc_b = (const float*)d_in[6];
  float* out = (float*)d_out;

  const int B = in_sizes[0] / TT;              // x is [B, TT, 1]
  const int grid = B / MROWS;                  // 2048/16 = 128 workgroups

  const size_t smem = (size_t)G4 * KST * 2     // Wsh
                    + (size_t)MROWS * GST * 4  // Gsh
                    + (size_t)MROWS * KST * 2  // Hh
                    + (size_t)MROWS * HID * 4  // Hf
                    + (size_t)MROWS * 4;       // Xsh  (~145.5 KB)

  (void)hipFuncSetAttribute((const void*)lstm_wmma_kernel,
                            hipFuncAttributeMaxDynamicSharedMemorySize, (int)smem);

  lstm_wmma_kernel<<<grid, BLOCK, smem, stream>>>(x, W_ih, W_hh, b_ih, b_hh, fc_W, fc_b, out);
}
